// HyperbolicTemporalAttention_14104672600228
// MI455X (gfx1250) — matrix operand order repaired and checked by reading, not compiled, not run on hardware
//
#include <hip/hip_runtime.h>
#include <hip/hip_bf16.h>
#include <math.h>

// ---- problem constants (match reference) ----
#define BB 16
#define TT 243
#define JJ 17
#define DD 256      // spatial dim d
#define DP 257      // d+1 (hyperboloid)
#define WW 3
#define KWIN 7
#define BJ (BB * JJ)        // 272 rows of length T
#define CT 48               // t-chunk per block
#define KM 64               // k/v rows staged per block (CT + 16 halo pad)
#define QM 48
#define NCHUNK ((TT + CT - 1) / CT)   // 6

typedef __bf16 bf16_t;
typedef __attribute__((ext_vector_type(16))) __bf16 v16bf;
typedef __attribute__((ext_vector_type(8)))  __bf16 v8bf;
typedef __attribute__((ext_vector_type(8)))  float  v8f;

// ---------------------------------------------------------------------------
// Kernel 0: convert Wq/Wk/Wv (f32, row-major [N=256][K=256]) -> bf16 in ws.
// 384 KB total -> stays resident in the 192 MB L2 for the whole fused kernel.
// ---------------------------------------------------------------------------
__global__ void wcvt_kernel(const float* __restrict__ wq,
                            const float* __restrict__ wk,
                            const float* __restrict__ wv,
                            bf16_t* __restrict__ wbf) {
  int i = blockIdx.x * 256 + threadIdx.x;        // 0 .. 65535
  wbf[i]           = (bf16_t)wq[i];
  wbf[65536 + i]   = (bf16_t)wk[i];
  wbf[131072 + i]  = (bf16_t)wv[i];
}

// ---------------------------------------------------------------------------
// Kernel 1: Lorentz log-map at origin, spatial part, cast to bf16.
// One wave32 per (bj,t) row; lane handles 8 of 256 elements; shfl reduction.
// Output layout: spat[bj*T + t][256] (row-major, 512B rows).
// ---------------------------------------------------------------------------
__global__ void logmap_kernel(const float* __restrict__ x,
                              bf16_t* __restrict__ spat) {
  const int wave = threadIdx.x >> 5;
  const int lane = threadIdx.x & 31;
  const int r = blockIdx.x * 8 + wave;           // row = bj*T + t
  if (r >= BJ * TT) return;
  const int bj = r / TT, t = r - bj * TT;
  const int b = bj / JJ, j = bj - b * JJ;
  const float* row = x + (size_t)((b * TT + t) * JJ + j) * DP;

  float xv[8];
  float ss = 0.f;
#pragma unroll
  for (int q = 0; q < 8; ++q) {
    xv[q] = row[1 + lane * 8 + q];
    ss += xv[q] * xv[q];
  }
#pragma unroll
  for (int m = 16; m >= 1; m >>= 1) ss += __shfl_xor(ss, m, 32);

  const float xt = row[0];
  const float sc = acoshf(fmaxf(xt, 1.0f + 1e-7f)) / fmaxf(sqrtf(ss), 1e-7f);

  union { v8bf v; uint4 u; } o;
#pragma unroll
  for (int q = 0; q < 8; ++q) o.v[q] = (bf16_t)(sc * xv[q]);
  *(uint4*)(spat + (size_t)r * DD + lane * 8) = o.u;   // 16B aligned
}

// ---------------------------------------------------------------------------
// Kernel 2: fused qkv-GEMM (WMMA bf16) + banded attention + exp-map.
// Block = 256 threads (8 waves), one (bj, t-chunk) per block.
// LDS: sA bf16[64][256] (32KB) + qS f32[48][256] + kS,vS f32[64][256]
//      = 212,992 B dynamic LDS (one block per 320KB WGP).
// Staging uses CDNA5 async global->LDS DMA (no VGPR round trip, ASYNCcnt).
// ---------------------------------------------------------------------------
__global__ void __launch_bounds__(256, 1)
fused_attn_kernel(const bf16_t* __restrict__ spat,
                  const bf16_t* __restrict__ wbf,
                  const float* __restrict__ bq,
                  const float* __restrict__ bk,
                  const float* __restrict__ bv,
                  const float* __restrict__ tau,
                  float* __restrict__ out) {
  extern __shared__ char smem[];
  bf16_t* sA = (bf16_t*)smem;                                  // [KM][DD]
  float*  qS = (float*)(smem + (size_t)KM * DD * sizeof(bf16_t)); // [QM][DD]
  float*  kS = qS + QM * DD;                                   // [KM][DD]
  float*  vS = kS + KM * DD;                                   // [KM][DD]

  const int bj = blockIdx.x;
  const int t0 = blockIdx.y * CT;
  const int kvBase = t0 - 8;      // sA row i  <->  t = kvBase + i (clamped)
  const int tid = threadIdx.x;
  const int lane = tid & 31, wave = tid >> 5;

  // ---- stage spatial rows [kvBase, kvBase+64) into LDS (edge-clamped) ----
  // CDNA5 async copy: data flows cache->LDS directly, tracked by ASYNCcnt.
  for (int idx = tid; idx < KM * (DD / 8); idx += 256) {
    const int rrow = idx >> 5;        // 32 x 16B chunks per 512B row
    const int chunk = idx & 31;
    int tsrc = kvBase + rrow;
    tsrc = tsrc < 0 ? 0 : (tsrc >= TT ? TT - 1 : tsrc);
    const bf16_t* gsrc = spat + ((size_t)bj * TT + tsrc) * DD + chunk * 8;
    bf16_t* ldst = sA + (size_t)rrow * DD + chunk * 8;
    // Low 32 bits of a generic LDS pointer == LDS byte address (flat aperture).
    unsigned lofs = (unsigned)(size_t)ldst;
    unsigned long long ga = (unsigned long long)(size_t)gsrc;
    asm volatile("global_load_async_to_lds_b128 %0, %1, off"
                 :: "v"(lofs), "v"(ga) : "memory");
  }
  asm volatile("s_wait_asynccnt 0x0" ::: "memory");
  __syncthreads();

  // ---- WMMA GEMM: 176 tiles of 16x16, K=256 (q:48, k:64, v:64) ----
  const int hh = lane >> 4;    // half-wave select per ISA bf16 A/B layouts
  const int ln = lane & 15;
  for (int tile = wave; tile < 176; tile += 8) {
    int mat, mtile, ntile, rowbase; float* dst; const float* bias;
    if (tile < 48)       { mat = 0; mtile = tile / 16;         ntile = tile % 16;         rowbase = 8 + mtile * 16; dst = qS; bias = bq; }
    else if (tile < 112) { mat = 1; mtile = (tile - 48) / 16;  ntile = (tile - 48) % 16;  rowbase = mtile * 16;     dst = kS; bias = bk; }
    else                 { mat = 2; mtile = (tile - 112) / 16; ntile = (tile - 112) % 16; rowbase = mtile * 16;     dst = vS; bias = bv; }

    // A: row M = rowbase+ln of sA; lanes 0-15 hold K {0-7,16-23}, 16-31 hold K {8-15,24-31}
    const bf16_t* arow = sA + (size_t)(rowbase + ln) * DD + hh * 8;
    // B: column N = W row (q = spatial @ W^T); lanes 0-15 hold K {0-7,16-23}, 16-31 hold K {8-15,24-31}
    const bf16_t* brow = wbf + ((size_t)mat << 16) + (size_t)(ntile * 16 + ln) * DD + hh * 8;
    __builtin_prefetch(brow, 0, 0);   // global_prefetch_b8 -> warm L2 line

    v8f c = {};
#pragma unroll
    for (int s = 0; s < 8; ++s) {     // K = 256 in steps of 32
      union { v16bf v; v8bf h[2]; } a, bm;
      a.h[0] = *(const v8bf*)(arow + s * 32);
      a.h[1] = *(const v8bf*)(arow + s * 32 + 16);
      bm.h[0] = *(const v8bf*)(brow + s * 32);
      bm.h[1] = *(const v8bf*)(brow + s * 32 + 16);
      c = __builtin_amdgcn_wmma_f32_16x16x32_bf16(
              /*neg_a=*/false, a.v, /*neg_b=*/false, bm.v,
              /*c_mod=*/(short)0, c, /*reuse_a=*/false, /*reuse_b=*/false);
    }
    // C layout: VGPR r -> M = r + 8*hh, N = ln.  Add bias, spill to LDS f32.
    const float bn = bias[ntile * 16 + ln];
    const int colb = ntile * 16 + ln;
#pragma unroll
    for (int r = 0; r < 8; ++r) {
      const int mrow = mtile * 16 + r + hh * 8;
      dst[mrow * DD + colb] = c[r] + bn;
    }
  }
  __syncthreads();

  // ---- banded softmax attention (KW=7) + Lorentz exp-map, one wave per t ----
  const float tauc = fmaxf(tau[0], 1e-3f);
  const float inv_scale = 1.0f / (16.0f * tauc);   // sqrt(256) = 16
  const int b = bj / JJ, jj = bj - b * JJ;

  for (int i = wave; i < CT; i += 8) {
    const int t = t0 + i;
    if (t >= TT) break;
    const int d0 = lane * 8;
    const float* qrow = qS + (size_t)i * DD + d0;
    float qv[8];
#pragma unroll
    for (int q = 0; q < 8; ++q) qv[q] = qrow[q];

    float lg[KWIN];
#pragma unroll
    for (int w = 0; w < KWIN; ++w) {
      const int tp = t + w - WW;
      if (tp < 0 || tp >= TT) { lg[w] = -3.0e38f; continue; }  // boundary mask
      const float* krow = kS + (size_t)(i + w + 5) * DD + d0;  // (tp - kvBase)
      float p = 0.f;
#pragma unroll
      for (int q = 0; q < 8; ++q) p += qv[q] * krow[q];
#pragma unroll
      for (int m = 16; m >= 1; m >>= 1) p += __shfl_xor(p, m, 32);
      lg[w] = p * inv_scale;
    }

    float mx = lg[0];
#pragma unroll
    for (int w = 1; w < KWIN; ++w) mx = fmaxf(mx, lg[w]);
    float pw[KWIN], sum = 0.f;
#pragma unroll
    for (int w = 0; w < KWIN; ++w) {
      pw[w] = (lg[w] > -1.0e38f) ? __expf(lg[w] - mx) : 0.f;
      sum += pw[w];
    }
    const float inv = 1.0f / sum;

    float acc[8] = {0.f, 0.f, 0.f, 0.f, 0.f, 0.f, 0.f, 0.f};
#pragma unroll
    for (int w = 0; w < KWIN; ++w) {
      if (pw[w] > 0.f) {
        const float* vrow = vS + (size_t)(i + w + 5) * DD + d0;
#pragma unroll
        for (int q = 0; q < 8; ++q) acc[q] += pw[w] * vrow[q];
      }
    }
    float ss = 0.f;
#pragma unroll
    for (int q = 0; q < 8; ++q) { acc[q] *= inv; ss += acc[q] * acc[q]; }
#pragma unroll
    for (int m = 16; m >= 1; m >>= 1) ss += __shfl_xor(ss, m, 32);

    const float n  = fmaxf(sqrtf(ss), 1e-7f);
    const float e  = __expf(n), ei = 1.0f / e;
    const float ch = 0.5f * (e + ei);
    const float sc = 0.5f * (e - ei) / n;          // sinh(n)/n

    float* orow = out + (size_t)((b * TT + t) * JJ + jj) * DP;
    if (lane == 0) orow[0] = ch;
#pragma unroll
    for (int q = 0; q < 8; ++q) orow[1 + d0 + q] = sc * acc[q];
  }
}

// ---------------------------------------------------------------------------
// host launch
// ---------------------------------------------------------------------------
extern "C" void kernel_launch(void* const* d_in, const int* in_sizes, int n_in,
                              void* d_out, int out_size, void* d_ws, size_t ws_size,
                              hipStream_t stream) {
  const float* x_seq = (const float*)d_in[0];
  // d_in[1] = vel_seq : unused by reference
  const float* tau = (const float*)d_in[2];
  const float* Wq = (const float*)d_in[3];
  const float* bq = (const float*)d_in[4];
  const float* Wk = (const float*)d_in[5];
  const float* bk = (const float*)d_in[6];
  const float* Wv = (const float*)d_in[7];
  const float* bv = (const float*)d_in[8];
  float* out = (float*)d_out;

  // workspace layout: [wbf bf16 3*256*256][spatial bf16 (BJ*T)*256]
  bf16_t* wbf  = (bf16_t*)d_ws;
  bf16_t* spat = wbf + 3 * 256 * 256;

  wcvt_kernel<<<256, 256, 0, stream>>>(Wq, Wk, Wv, wbf);

  const int nrows = BJ * TT;                       // 66096 (divisible by 8)
  logmap_kernel<<<nrows / 8, 256, 0, stream>>>(x_seq, spat);

  const size_t lds_bytes = (size_t)KM * DD * sizeof(bf16_t)
                         + (size_t)(QM + KM + KM) * DD * sizeof(float); // 212,992
  dim3 grid(BJ, NCHUNK);
  fused_attn_kernel<<<grid, 256, lds_bytes, stream>>>(spat, wbf, bq, bk, bv, tau, out);
}
